// MultiHeadAttention_32985348833534
// MI455X (gfx1250) — matrix operand order repaired and checked by reading, not compiled, not run on hardware
//
#include <hip/hip_runtime.h>
#include <stdint.h>

typedef __bf16 v16bf __attribute__((ext_vector_type(16)));
typedef float  v8f   __attribute__((ext_vector_type(8)));
typedef int    v4i   __attribute__((ext_vector_type(4)));

struct __attribute__((aligned(16))) U4 { unsigned int x, y, z, w; };

union Frag {
  v16bf v;
  U4 q[2];
  unsigned short s[16];
};

#define WMMA_BF16(A, B, C) \
  __builtin_amdgcn_wmma_f32_16x16x32_bf16(false, (A), false, (B), (short)0, (C), false, false)

// ---- CDNA5 async global->LDS (ASYNCcnt-tracked) if the toolchain exposes it
#if defined(__has_builtin)
#if __has_builtin(__builtin_amdgcn_global_load_async_to_lds_b128)
#define HAVE_ASYNC_LDS 1
#endif
#endif

#ifdef HAVE_ASYNC_LDS
#define AS1 __attribute__((address_space(1)))
#define AS3 __attribute__((address_space(3)))
__device__ __forceinline__ void async_b128(const unsigned short* g, unsigned short* l) {
  __builtin_amdgcn_global_load_async_to_lds_b128((AS1 v4i*)g, (AS3 v4i*)l, 0, 0);
}
__device__ __forceinline__ void wait_async0() {
#if __has_builtin(__builtin_amdgcn_s_wait_asynccnt)
  __builtin_amdgcn_s_wait_asynccnt(0);
#else
  asm volatile("s_wait_asynccnt 0" ::: "memory");
#endif
}
#endif

__device__ __forceinline__ unsigned short f2bf(float f) {
  unsigned int u = __float_as_uint(f);
  u += 0x7FFFu + ((u >> 16) & 1u);           // round-to-nearest-even
  return (unsigned short)(u >> 16);
}

__device__ __forceinline__ float lane_bcast(float v, int srcLane) {
  return __int_as_float(__builtin_amdgcn_ds_bpermute(srcLane << 2, __float_as_int(v)));
}

// ---------------------------------------------------------------- convert
__global__ void cvt_f32_to_bf16(const float* __restrict__ in,
                                unsigned short* __restrict__ out, int n) {
  int i = blockIdx.x * blockDim.x + threadIdx.x;
  if (i < n) out[i] = f2bf(in[i]);
}

// ---------------------------------------------------------------- projection GEMM
// out[M=4096, N=1024] = A[4096,1024](bf16) * W^T (W row-major [1024,1024] bf16) + bias
// One wave computes a 32x64 tile. MODE 0: bf16 row-major out.
// MODE 1: f32 row-major out. MODE 2: bf16 per-head-transposed out Vt[bh][dk][s].
template<int MODE>
__global__ void __launch_bounds__(256) gemm_xwT(
    const unsigned short* __restrict__ A,
    const unsigned short* __restrict__ W,
    const float* __restrict__ bias,
    void* __restrict__ outp)
{
  const int lane = threadIdx.x & 31;
  const int wid  = threadIdx.x >> 5;
  const int wg   = blockIdx.x * 8 + wid;     // 2048 wave-tiles
  const int tm   = wg >> 4;                  // 128 M-tiles of 32
  const int tn   = wg & 15;                  // 16 N-tiles of 64
  const int m0 = tm * 32, n0 = tn * 64;
  const int ln = lane & 15;
  const int hi = lane >> 4;

  v8f acc[2][4];
#pragma unroll
  for (int i = 0; i < 2; ++i)
#pragma unroll
    for (int j = 0; j < 4; ++j) acc[i][j] = {};

  for (int kc = 0; kc < 1024; kc += 32) {
    Frag a[2];
#pragma unroll
    for (int mi = 0; mi < 2; ++mi) {
      const unsigned short* p = A + (size_t)(m0 + mi * 16 + ln) * 1024 + kc + hi * 8;
      a[mi].q[0] = *(const U4*)p;
      a[mi].q[1] = *(const U4*)(p + 16);
    }
    Frag b[4];
#pragma unroll
    for (int ni = 0; ni < 4; ++ni) {
      const unsigned short* p = W + (size_t)(n0 + ni * 16 + ln) * 1024 + kc + hi * 8;
      b[ni].q[0] = *(const U4*)p;
      b[ni].q[1] = *(const U4*)(p + 16);
    }
#pragma unroll
    for (int mi = 0; mi < 2; ++mi)
#pragma unroll
      for (int ni = 0; ni < 4; ++ni)
        acc[mi][ni] = WMMA_BF16(a[mi].v, b[ni].v, acc[mi][ni]);
  }

#pragma unroll
  for (int ni = 0; ni < 4; ++ni) {
    const int n = n0 + ni * 16 + ln;
    const float bn = bias[n];
#pragma unroll
    for (int mi = 0; mi < 2; ++mi) {
#pragma unroll
      for (int r = 0; r < 8; ++r) {
        const int mrow = m0 + mi * 16 + r + hi * 8;
        const float val = acc[mi][ni][r] + bn;
        if (MODE == 0) {
          ((unsigned short*)outp)[(size_t)mrow * 1024 + n] = f2bf(val);
        } else if (MODE == 1) {
          ((float*)outp)[(size_t)mrow * 1024 + n] = val;
        } else {
          const int bb = mrow >> 11, ss = mrow & 2047;
          const int hh = n >> 6,    dk = n & 63;
          ((unsigned short*)outp)[((size_t)(bb * 16 + hh) * 64 + dk) * 2048 + ss] = f2bf(val);
        }
      }
    }
  }
}

// ---------------------------------------------------------------- fused flash attention
// Workgroup = 8 waves, all on the same (b,h); wave w handles query tile qb*8+w.
// K/V 64-key chunks are staged cooperatively into LDS (double-buffered), cutting
// global traffic 8x. Scores computed transposed (S^T = K x Q^T) so exp'd tiles
// ARE the A-fragment of the P*V WMMA (no transpose anywhere).
__global__ void __launch_bounds__(256) attn_fused(
    const unsigned short* __restrict__ Qp,   // [4096,1024] bf16
    const unsigned short* __restrict__ Kp,   // [4096,1024] bf16
    const unsigned short* __restrict__ Vt,   // [32][64][2048] bf16 (per-head V^T)
    unsigned short* __restrict__ ctx)        // [4096,1024] bf16
{
  __shared__ unsigned short Ksh[2][64 * 72]; // 64 keys x 64 d, stride 72 (bank-safe)
  __shared__ unsigned short Vsh[2][64 * 72]; // 64 d x 64 keys, stride 72

  const int tid  = threadIdx.x;
  const int lane = tid & 31;
  const int wid  = tid >> 5;
  const int bh = blockIdx.x >> 4;            // 0..31
  const int qb = blockIdx.x & 15;
  const int b = bh >> 4, h = bh & 15;
  const int ln = lane & 15, hi = lane >> 4;
  const int q0 = (qb * 8 + wid) * 16;

  const unsigned short* Kbase = Kp + (size_t)(b * 2048) * 1024 + h * 64;
  const unsigned short* Vbase = Vt + (size_t)bh * 64 * 2048;

  // cooperative staging: thread owns two b128 segments (rows r0 and r0+32)
  const int r0 = tid >> 3;                   // 0..31
  const int c0 = (tid & 7) * 8;              // 0..56 (shorts)

  // Q fragments (B operand of S^T GEMM): loop invariant.
  Frag qf[2];
  {
    const unsigned short* qrow = Qp + (size_t)(b * 2048 + q0 + ln) * 1024 + h * 64;
#pragma unroll
    for (int d = 0; d < 2; ++d) {
      const unsigned short* p = qrow + d * 32 + hi * 8;
      qf[d].q[0] = *(const U4*)p;
      qf[d].q[1] = *(const U4*)(p + 16);
    }
  }

  v8f acc[4];
#pragma unroll
  for (int t = 0; t < 4; ++t) acc[t] = {};
  float mrun = -1e30f, lrun = 0.f;

  auto compute_chunk = [&](const unsigned short* Kt, const unsigned short* Vl) {
#pragma unroll
    for (int sub = 0; sub < 64; sub += 32) {
      v8f st[2];
#pragma unroll
      for (int t = 0; t < 2; ++t) {
        v8f s = {};
        const unsigned short* kr = Kt + (sub + t * 16 + ln) * 72;
#pragma unroll
        for (int d = 0; d < 2; ++d) {
          Frag af;
          const unsigned short* p = kr + d * 32 + hi * 8;
          af.q[0] = *(const U4*)p;
          af.q[1] = *(const U4*)(p + 16);
          s = WMMA_BF16(af.v, qf[d].v, s);
        }
        st[t] = s;
      }

      // online softmax: lane owns query q=lane%16; partner lane^16 has the
      // other 16 key values of this 32-key sub-chunk.
      float cm = -1e30f;
#pragma unroll
      for (int t = 0; t < 2; ++t)
#pragma unroll
        for (int r = 0; r < 8; ++r) {
          st[t][r] *= 0.125f;                // 1/sqrt(64)
          cm = fmaxf(cm, st[t][r]);
        }
      cm = fmaxf(cm, lane_bcast(cm, lane ^ 16));
      const float mnew  = fmaxf(mrun, cm);
      const float alpha = __expf(mrun - mnew);
      mrun = mnew;

      Frag pf;                               // exp'd scores == A-fragment layout
      float rs = 0.f;
#pragma unroll
      for (int t = 0; t < 2; ++t)
#pragma unroll
        for (int r = 0; r < 8; ++r) {
          const float pv = __expf(st[t][r] - mnew);
          rs += pv;
          pf.s[t * 8 + r] = f2bf(pv);
        }
      rs += lane_bcast(rs, lane ^ 16);
      lrun = lrun * alpha + rs;

#pragma unroll
      for (int r = 0; r < 8; ++r) {
        const float ar = lane_bcast(alpha, r + hi * 8);
#pragma unroll
        for (int t = 0; t < 4; ++t) acc[t][r] *= ar;
      }

#pragma unroll
      for (int t = 0; t < 4; ++t) {
        Frag vf;
        const unsigned short* p = Vl + (t * 16 + ln) * 72 + sub + hi * 8;
        vf.q[0] = *(const U4*)p;
        vf.q[1] = *(const U4*)(p + 16);
        acc[t] = WMMA_BF16(pf.v, vf.v, acc[t]);
      }
    }
  };

#ifdef HAVE_ASYNC_LDS
  // ---- async (ASYNCcnt) double-buffered pipeline: DMA chunk j+1 while
  // computing chunk j; s_wait_asynccnt 0 + barrier publishes chunk j.
#pragma unroll
  for (int i = 0; i < 2; ++i) {
    const int r = r0 + i * 32;
    async_b128(Kbase + (size_t)r * 1024 + c0, &Ksh[0][r * 72 + c0]);
    async_b128(Vbase + (size_t)r * 2048 + c0, &Vsh[0][r * 72 + c0]);
  }
  for (int j = 0; j < 32; ++j) {
    const int cur = j & 1;
    wait_async0();
    __syncthreads();
    if (j + 1 < 32) {
      const int kc = (j + 1) * 64;
#pragma unroll
      for (int i = 0; i < 2; ++i) {
        const int r = r0 + i * 32;
        async_b128(Kbase + (size_t)(kc + r) * 1024 + c0, &Ksh[1 - cur][r * 72 + c0]);
        async_b128(Vbase + (size_t)r * 2048 + kc + c0, &Vsh[1 - cur][r * 72 + c0]);
      }
    }
    compute_chunk(&Ksh[cur][0], &Vsh[cur][0]);
  }
#else
  // ---- fallback: prefetch chunk j+1 into VGPRs during compute of chunk j.
  U4 rk[2], rv[2];
#pragma unroll
  for (int i = 0; i < 2; ++i) {
    const int r = r0 + i * 32;
    rk[i] = *(const U4*)(Kbase + (size_t)r * 1024 + c0);
    rv[i] = *(const U4*)(Vbase + (size_t)r * 2048 + c0);
  }
  for (int j = 0; j < 32; ++j) {
    const int cur = j & 1;
#pragma unroll
    for (int i = 0; i < 2; ++i) {
      const int r = r0 + i * 32;
      *(U4*)&Ksh[cur][r * 72 + c0] = rk[i];
      *(U4*)&Vsh[cur][r * 72 + c0] = rv[i];
    }
    __syncthreads();
    if (j + 1 < 32) {
      const int kc = (j + 1) * 64;
#pragma unroll
      for (int i = 0; i < 2; ++i) {
        const int r = r0 + i * 32;
        rk[i] = *(const U4*)(Kbase + (size_t)(kc + r) * 1024 + c0);
        rv[i] = *(const U4*)(Vbase + (size_t)r * 2048 + kc + c0);
      }
    }
    compute_chunk(&Ksh[cur][0], &Vsh[cur][0]);
    __syncthreads();
  }
#endif

  const float inv = 1.0f / lrun;
#pragma unroll
  for (int r = 0; r < 8; ++r) {
    const float ir = lane_bcast(inv, r + hi * 8);
    const int mrow = b * 2048 + q0 + r + hi * 8;
#pragma unroll
    for (int t = 0; t < 4; ++t)
      ctx[(size_t)mrow * 1024 + h * 64 + t * 16 + ln] = f2bf(acc[t][r] * ir);
  }
}

// ---------------------------------------------------------------- launch
extern "C" void kernel_launch(void* const* d_in, const int* in_sizes, int n_in,
                              void* d_out, int out_size, void* d_ws, size_t ws_size,
                              hipStream_t stream) {
  (void)in_sizes; (void)n_in; (void)out_size; (void)ws_size;

  const float* query = (const float*)d_in[0];
  const float* key   = (const float*)d_in[1];
  const float* value = (const float*)d_in[2];
  const float* Wq = (const float*)d_in[3];
  const float* bq = (const float*)d_in[4];
  const float* Wk = (const float*)d_in[5];
  const float* bk = (const float*)d_in[6];
  const float* Wv = (const float*)d_in[7];
  const float* bv = (const float*)d_in[8];
  const float* Wo = (const float*)d_in[9];
  const float* bo = (const float*)d_in[10];

  const size_t ACT = (size_t)4096 * 1024;   // activation elements
  const size_t WEL = (size_t)1024 * 1024;   // weight elements

  char* ws = (char*)d_ws;
  size_t off = 0;
  auto take = [&](size_t elems) -> unsigned short* {
    unsigned short* p = (unsigned short*)(ws + off);
    off += ((elems * 2) + 255) & ~(size_t)255;
    return p;
  };
  unsigned short* qb  = take(ACT);
  unsigned short* kb  = take(ACT);
  unsigned short* vb  = take(ACT);
  unsigned short* wqb = take(WEL);
  unsigned short* wkb = take(WEL);
  unsigned short* wvb = take(WEL);
  unsigned short* wob = take(WEL);
  unsigned short* Qp  = take(ACT);
  unsigned short* Kp  = take(ACT);
  unsigned short* Vtw = take(ACT);
  unsigned short* ctx = take(ACT);

  const int CT = 256;
  cvt_f32_to_bf16<<<(int)((ACT + CT - 1) / CT), CT, 0, stream>>>(query, qb, (int)ACT);
  cvt_f32_to_bf16<<<(int)((ACT + CT - 1) / CT), CT, 0, stream>>>(key,   kb, (int)ACT);
  cvt_f32_to_bf16<<<(int)((ACT + CT - 1) / CT), CT, 0, stream>>>(value, vb, (int)ACT);
  cvt_f32_to_bf16<<<(int)((WEL + CT - 1) / CT), CT, 0, stream>>>(Wq, wqb, (int)WEL);
  cvt_f32_to_bf16<<<(int)((WEL + CT - 1) / CT), CT, 0, stream>>>(Wk, wkb, (int)WEL);
  cvt_f32_to_bf16<<<(int)((WEL + CT - 1) / CT), CT, 0, stream>>>(Wv, wvb, (int)WEL);
  cvt_f32_to_bf16<<<(int)((WEL + CT - 1) / CT), CT, 0, stream>>>(Wo, wob, (int)WEL);

  // 2048 wave-tiles / 8 waves per 256-thread block = 256 blocks
  gemm_xwT<0><<<256, 256, 0, stream>>>(qb, wqb, bq, Qp);
  gemm_xwT<0><<<256, 256, 0, stream>>>(kb, wkb, bk, Kp);
  gemm_xwT<2><<<256, 256, 0, stream>>>(vb, wvb, bv, Vtw);

  // 32 (b,h) x 16 q-blocks = 512 blocks; 8 q-tiles per block
  attn_fused<<<512, 256, 0, stream>>>(Qp, Kp, Vtw, ctx);

  gemm_xwT<1><<<256, 256, 0, stream>>>(ctx, wob, bo, d_out);
}
